// MultiScaleDeformableAttention_55336358642083
// MI455X (gfx1250) — matrix operand order repaired and checked by reading, not compiled, not run on hardware
//
#include <hip/hip_runtime.h>
#include <math.h>

typedef __attribute__((ext_vector_type(16))) _Float16 v16h;
typedef __attribute__((ext_vector_type(8)))  _Float16 v8h;
typedef __attribute__((ext_vector_type(4)))  _Float16 v4h;
typedef __attribute__((ext_vector_type(8)))  float    v8f;

#define EMBED    256
#define HEADS    8
#define HEAD_DIM 32
#define LEVELS   4
#define POINTS   4

// ---------------------------------------------------------------------------
// Elementwise f32 -> f16 (n must be a multiple of 4; vectorized 4/thread)
// ---------------------------------------------------------------------------
__global__ void msda_f32_to_f16(const float* __restrict__ src,
                                _Float16* __restrict__ dst, int n4) {
    int i = blockIdx.x * blockDim.x + threadIdx.x;
    if (i >= n4) return;
    const float4 v = ((const float4*)src)[i];
    v4h o;
    o[0] = (_Float16)v.x; o[1] = (_Float16)v.y;
    o[2] = (_Float16)v.z; o[3] = (_Float16)v.w;
    ((v4h*)dst)[i] = o;
}

// ---------------------------------------------------------------------------
// Convert + transpose weights: dst(Nout, K) f16  <-  src(K, Nout) f32
// ---------------------------------------------------------------------------
__global__ void msda_convert_wT(const float* __restrict__ src,
                                _Float16* __restrict__ dst,
                                int K, int Nout) {
    int idx = blockIdx.x * blockDim.x + threadIdx.x;
    if (idx >= K * Nout) return;
    int k = idx / Nout;
    int n = idx - k * Nout;
    dst[(size_t)n * K + k] = (_Float16)src[idx];
}

// ---------------------------------------------------------------------------
// Y(M,Nout) = Xh(M,K) @ W(K,Nout) + b    (Xh in f16, W pre-transposed f16)
// One wave computes a 32x64 output tile: two A-fragments (rows r..r+15 and
// r+16..r+31) share four B-fragments -> 8 v_wmma_f32_16x16x32_f16 per K-step,
// 1.5 b128 loads per WMMA. K = 256 -> 64 WMMAs per wave.
// ---------------------------------------------------------------------------
__global__ void msda_gemm_wmma(const _Float16* __restrict__ Xh,
                               const _Float16* __restrict__ WhT,
                               const float* __restrict__ bias,
                               float* __restrict__ Y,
                               int M, int K, int Nout) {
    const int lane    = threadIdx.x & 31;
    const int wpb     = blockDim.x >> 5;
    const int tile    = blockIdx.x * wpb + (threadIdx.x >> 5);
    const int groupsN = Nout >> 6;              // 64 output columns per wave
    const int tilesM  = (M + 31) >> 5;          // 32 output rows per wave
    if (tile >= tilesM * groupsN) return;
    const int gn = tile % groupsN;
    const int tm = tile / groupsN;

    const int half = lane >> 4;                 // 0: lanes 0-15, 1: lanes 16-31
    const int l15  = lane & 15;

    // Two A-fragment rows (clamped for ragged last tile; stores are guarded)
    int aRow0 = tm * 32 + l15;       if (aRow0 >= M) aRow0 = M - 1;
    int aRow1 = tm * 32 + 16 + l15;  if (aRow1 >= M) aRow1 = M - 1;
    const _Float16* xrow0 = Xh + (size_t)aRow0 * K;
    const _Float16* xrow1 = Xh + (size_t)aRow1 * K;

    // A 16x32 layout (matches B): lane<16 holds K 0..15; lane>=16 holds 16..31
    const int kHi  = half * 8;
    // B 32x16 layout used here (contiguous k): lane<16 holds K 0..15 of its column; lane>=16 holds 16..31
    const int kOff = half * 16;

    const int col0 = gn * 64 + l15;
    const _Float16* wc0 = WhT + (size_t)(col0)      * K;
    const _Float16* wc1 = WhT + (size_t)(col0 + 16) * K;
    const _Float16* wc2 = WhT + (size_t)(col0 + 32) * K;
    const _Float16* wc3 = WhT + (size_t)(col0 + 48) * K;

    v8f a00 = {}, a01 = {}, a02 = {}, a03 = {};   // rows 0..15 of the tile
    v8f a10 = {}, a11 = {}, a12 = {}, a13 = {};   // rows 16..31 of the tile
    for (int kb = 0; kb < K; kb += 32) {
        const v8h p0 = *(const v8h*)(xrow0 + kb + kOff);        // 16B aligned
        const v8h p1 = *(const v8h*)(xrow0 + kb + kOff + 8);
        const v8h q0 = *(const v8h*)(xrow1 + kb + kOff);
        const v8h q1 = *(const v8h*)(xrow1 + kb + kOff + 8);
        const v16h aF0 = __builtin_shufflevector(p0, p1,
                            0, 1, 2, 3, 4, 5, 6, 7, 8, 9, 10, 11, 12, 13, 14, 15);
        const v16h aF1 = __builtin_shufflevector(q0, q1,
                            0, 1, 2, 3, 4, 5, 6, 7, 8, 9, 10, 11, 12, 13, 14, 15);
        const v16h b0 = *(const v16h*)(wc0 + kb + kOff);       // 32B aligned
        const v16h b1 = *(const v16h*)(wc1 + kb + kOff);
        const v16h b2 = *(const v16h*)(wc2 + kb + kOff);
        const v16h b3 = *(const v16h*)(wc3 + kb + kOff);
        a00 = __builtin_amdgcn_wmma_f32_16x16x32_f16(false, aF0, false, b0, (short)0, a00, false, false);
        a01 = __builtin_amdgcn_wmma_f32_16x16x32_f16(false, aF0, false, b1, (short)0, a01, false, false);
        a02 = __builtin_amdgcn_wmma_f32_16x16x32_f16(false, aF0, false, b2, (short)0, a02, false, false);
        a03 = __builtin_amdgcn_wmma_f32_16x16x32_f16(false, aF0, false, b3, (short)0, a03, false, false);
        a10 = __builtin_amdgcn_wmma_f32_16x16x32_f16(false, aF1, false, b0, (short)0, a10, false, false);
        a11 = __builtin_amdgcn_wmma_f32_16x16x32_f16(false, aF1, false, b1, (short)0, a11, false, false);
        a12 = __builtin_amdgcn_wmma_f32_16x16x32_f16(false, aF1, false, b2, (short)0, a12, false, false);
        a13 = __builtin_amdgcn_wmma_f32_16x16x32_f16(false, aF1, false, b3, (short)0, a13, false, false);
    }

    // C/D layout: VGPR v -> M = v (lanes 0-15) or 8+v (lanes 16-31); column = lane&15
    const float bv0 = bias[col0];
    const float bv1 = bias[col0 + 16];
    const float bv2 = bias[col0 + 32];
    const float bv3 = bias[col0 + 48];
    const int rowBase0 = tm * 32 + half * 8;
    const int rowBase1 = rowBase0 + 16;
#pragma unroll
    for (int v = 0; v < 8; ++v) {
        int r0 = rowBase0 + v;
        if (r0 < M) {
            float* yr = Y + (size_t)r0 * Nout;
            yr[col0]      = a00[v] + bv0;
            yr[col0 + 16] = a01[v] + bv1;
            yr[col0 + 32] = a02[v] + bv2;
            yr[col0 + 48] = a03[v] + bv3;
        }
        int r1 = rowBase1 + v;
        if (r1 < M) {
            float* yr = Y + (size_t)r1 * Nout;
            yr[col0]      = a10[v] + bv0;
            yr[col0 + 16] = a11[v] + bv1;
            yr[col0 + 32] = a12[v] + bv2;
            yr[col0 + 48] = a13[v] + bv3;
        }
    }
}

// ---------------------------------------------------------------------------
// In-place softmax over groups of 16 (per n,q,head). buf layout: (total, 16)
// ---------------------------------------------------------------------------
__global__ void msda_softmax16(float* __restrict__ buf, int total) {
    int i = blockIdx.x * blockDim.x + threadIdx.x;
    if (i >= total) return;
    float* x = buf + (size_t)i * 16;
    float m = x[0];
#pragma unroll
    for (int k = 1; k < 16; ++k) m = fmaxf(m, x[k]);
    float e[16];
    float s = 0.f;
#pragma unroll
    for (int k = 0; k < 16; ++k) { e[k] = __expf(x[k] - m); s += e[k]; }
    float inv = 1.f / s;
#pragma unroll
    for (int k = 0; k < 16; ++k) x[k] = e[k] * inv;
}

// ---------------------------------------------------------------------------
// Deformable sampling core. One wave per (n, q, head); lane = head-dim channel.
// val layout: (N, Lv, HEADS*HEAD_DIM) f32 — gathers are 128B coalesced / wave,
// served from L2 (41 MB tensor << 192 MB L2). Output written directly in f16
// to feed the final WMMA GEMM with no extra conversion pass.
// ---------------------------------------------------------------------------
__global__ void msda_sample(const float* __restrict__ off,
                            const float* __restrict__ aw,
                            const float* __restrict__ refp,
                            const float* __restrict__ val,
                            _Float16* __restrict__ out,
                            int N, int Lq) {
    const int lvlH[LEVELS] = {100, 50, 25, 13};
    const int lvlW[LEVELS] = {152, 76, 38, 19};
    const int lvlO[LEVELS] = {0, 15200, 19000, 19950};

    const int lane  = threadIdx.x & 31;
    const int wave  = blockIdx.x * (blockDim.x >> 5) + (threadIdx.x >> 5);
    const int total = N * Lq * HEADS;
    if (wave >= total) return;
    const int h = wave % HEADS;
    const int t = wave / HEADS;          // t = n*Lq + q
    const int n = t / Lq;

    const float* offp = off  + (size_t)t * (HEADS * LEVELS * POINTS * 2) + h * (LEVELS * POINTS * 2);
    const float* awp  = aw   + (size_t)t * (HEADS * LEVELS * POINTS)     + h * (LEVELS * POINTS);
    const float* rp   = refp + (size_t)t * (LEVELS * 4);
    const float* vb   = val  + (size_t)n * Lq * EMBED + h * HEAD_DIM + lane;

    float acc = 0.f;
#pragma unroll
    for (int l = 0; l < LEVELS; ++l) {
        const int H = lvlH[l], W = lvlW[l], O = lvlO[l];
        const float cx = rp[l * 4 + 0], cy = rp[l * 4 + 1];
        const float rw = rp[l * 4 + 2], rh = rp[l * 4 + 3];
#pragma unroll
        for (int pt = 0; pt < POINTS; ++pt) {
            const float ox = offp[(l * POINTS + pt) * 2 + 0];
            const float oy = offp[(l * POINTS + pt) * 2 + 1];
            const float a  = awp[l * POINTS + pt];
            // loc = c + off/POINTS * wh * 0.5 ; grid = 2*loc-1 ; ix = (gx+1)*W/2 - 0.5
            const float ix = (cx + ox * rw * 0.125f) * (float)W - 0.5f;
            const float iy = (cy + oy * rh * 0.125f) * (float)H - 0.5f;
            const float fx0 = floorf(ix), fy0 = floorf(iy);
            const int   x0 = (int)fx0,   y0 = (int)fy0;
            const float wx1 = ix - fx0,  wy1 = iy - fy0;
            const float wx0 = 1.f - wx1, wy0 = 1.f - wy1;
            const bool vx0 = (x0 >= 0) && (x0 < W);
            const bool vx1 = (x0 + 1 >= 0) && (x0 + 1 < W);
            const bool vy0 = (y0 >= 0) && (y0 < H);
            const bool vy1 = (y0 + 1 >= 0) && (y0 + 1 < H);
            float s = 0.f;
            if (vx0 && vy0) s += wx0 * wy0 * vb[(size_t)(O + y0 * W + x0) * EMBED];
            if (vx1 && vy0) s += wx1 * wy0 * vb[(size_t)(O + y0 * W + x0 + 1) * EMBED];
            if (vx0 && vy1) s += wx0 * wy1 * vb[(size_t)(O + (y0 + 1) * W + x0) * EMBED];
            if (vx1 && vy1) s += wx1 * wy1 * vb[(size_t)(O + (y0 + 1) * W + x0 + 1) * EMBED];
            acc += a * s;
        }
    }
    out[(size_t)t * EMBED + h * HEAD_DIM + lane] = (_Float16)acc;
}

// ---------------------------------------------------------------------------
extern "C" void kernel_launch(void* const* d_in, const int* in_sizes, int n_in,
                              void* d_out, int out_size, void* d_ws, size_t ws_size,
                              hipStream_t stream) {
    (void)n_in; (void)out_size; (void)ws_size;

    const float* query  = (const float*)d_in[0];
    const float* refp   = (const float*)d_in[1];
    const float* value  = (const float*)d_in[2];
    // d_in[3] = spatial_shapes (int64) — shapes are fixed compile-time constants
    const float* W_off  = (const float*)d_in[4];
    const float* b_off  = (const float*)d_in[5];
    const float* W_attw = (const float*)d_in[6];
    const float* b_attw = (const float*)d_in[7];
    const float* W_val  = (const float*)d_in[8];
    const float* b_val  = (const float*)d_in[9];
    const float* W_out  = (const float*)d_in[10];
    const float* b_out  = (const float*)d_in[11];

    const int Lq = 15200 + 3800 + 950 + 247;        // 20197 (fixed by SHAPES)
    const int N  = in_sizes[0] / (Lq * EMBED);
    const int M  = N * Lq;

    // Workspace carve-up (256B aligned)
    char*  ws = (char*)d_ws;
    size_t p  = 0;
    auto take = [&](size_t bytes) -> char* {
        char* r = ws + p;
        p = (p + bytes + 255) & ~(size_t)255;
        return r;
    };
    _Float16* whOff = (_Float16*)take((size_t)EMBED * 256 * sizeof(_Float16));
    _Float16* whAtt = (_Float16*)take((size_t)EMBED * 128 * sizeof(_Float16));
    _Float16* whVal = (_Float16*)take((size_t)EMBED * 256 * sizeof(_Float16));
    _Float16* whOut = (_Float16*)take((size_t)EMBED * 256 * sizeof(_Float16));
    _Float16* qh    = (_Float16*)take((size_t)M * EMBED * sizeof(_Float16));
    _Float16* vh    = (_Float16*)take((size_t)M * EMBED * sizeof(_Float16));
    _Float16* midH  = (_Float16*)take((size_t)M * EMBED * sizeof(_Float16));
    float* offBuf = (float*)take((size_t)M * 256 * sizeof(float));
    float* attBuf = (float*)take((size_t)M * 128 * sizeof(float));
    float* valBuf = (float*)take((size_t)M * 256 * sizeof(float));

    const int cth = 256;

    // 1) Activations -> f16 (once), weights -> f16 transposed (once)
    const int n4 = (M * EMBED) / 4;
    msda_f32_to_f16<<<(n4 + cth - 1) / cth, cth, 0, stream>>>(query, qh, n4);
    msda_f32_to_f16<<<(n4 + cth - 1) / cth, cth, 0, stream>>>(value, vh, n4);
    msda_convert_wT<<<(EMBED * 256 + cth - 1) / cth, cth, 0, stream>>>(W_off,  whOff, EMBED, 256);
    msda_convert_wT<<<(EMBED * 128 + cth - 1) / cth, cth, 0, stream>>>(W_attw, whAtt, EMBED, 128);
    msda_convert_wT<<<(EMBED * 256 + cth - 1) / cth, cth, 0, stream>>>(W_val,  whVal, EMBED, 256);
    msda_convert_wT<<<(EMBED * 256 + cth - 1) / cth, cth, 0, stream>>>(W_out,  whOut, EMBED, 256);

    // 2) Projection GEMMs (WMMA, 32x64 tile per wave)
    const int tilesM = (M + 31) / 32;
    const int WPB    = 8;                 // 8 waves / 256-thread block
    auto blocksFor = [&](int groupsN) { return (tilesM * groupsN + WPB - 1) / WPB; };
    msda_gemm_wmma<<<blocksFor(4), 32 * WPB, 0, stream>>>(qh, whOff, b_off,  offBuf, M, EMBED, 256);
    msda_gemm_wmma<<<blocksFor(2), 32 * WPB, 0, stream>>>(qh, whAtt, b_attw, attBuf, M, EMBED, 128);
    msda_gemm_wmma<<<blocksFor(4), 32 * WPB, 0, stream>>>(vh, whVal, b_val,  valBuf, M, EMBED, 256);

    // 3) Softmax over the 16 (level,point) weights per (n,q,head), in place
    const int totalGroups = M * HEADS;
    msda_softmax16<<<(totalGroups + 255) / 256, 256, 0, stream>>>(attBuf, totalGroups);

    // 4) Bilinear sampling + attention-weighted sum -> f16 (feeds final GEMM)
    const int totalWaves = M * HEADS;
    msda_sample<<<(totalWaves + 7) / 8, 256, 0, stream>>>(offBuf, attBuf, refp, valBuf, midH, N, Lq);

    // 5) Output projection GEMM (WMMA) -> d_out
    msda_gemm_wmma<<<blocksFor(4), 32 * WPB, 0, stream>>>(midH, whOut, b_out, (float*)d_out, M, EMBED, 256);
}
